// FluxonUpdaterCos_15444702396964
// MI455X (gfx1250) — compile-verified, matrix operand order checked by reading, not run on hardware
//
#include <hip/hip_runtime.h>
#include <hip/hip_bf16.h>

typedef __attribute__((ext_vector_type(16))) __bf16 v16bf;
typedef __attribute__((ext_vector_type(8)))  float  v8f;

#define BB   131072
#define DD   256
#define HH   512
#define KK   8192
#define G3H  1536

// ---------------- phase 1: zero accumulators -------------------------------
__global__ void k_zero(float4* p, int n4) {
    int i = blockIdx.x * blockDim.x + threadIdx.x;
    if (i < n4) p[i] = make_float4(0.f, 0.f, 0.f, 0.f);
}

// ---------------- phase 2: segment scatter-add -----------------------------
// one block per batch row; 256 threads cover msg = [h_fast | h_slow]
__global__ void k_scatter(const float* __restrict__ hf, const float* __restrict__ hs,
                          const long long* __restrict__ idx,
                          float* __restrict__ sums, float* __restrict__ cnt) {
    int b = blockIdx.x;
    int t = threadIdx.x;                 // 0..255
    int seg = (int)idx[b];
    float v1 = hf[b * DD + t];
    float v2 = hs[b * DD + t];
    atomicAdd(sums + seg * HH + t,       v1);
    atomicAdd(sums + seg * HH + DD + t,  v2);
    if (t == 0) atomicAdd(cnt + seg, 1.0f);
}

// ---------------- phase 3: mean + split-bf16 conversions -------------------
__global__ void k_mean_split(const float* __restrict__ sums, const float* __restrict__ cnt,
                             __bf16* __restrict__ hi, __bf16* __restrict__ lo, int n) {
    int i = blockIdx.x * blockDim.x + threadIdx.x;
    if (i >= n) return;
    float m = sums[i] / (cnt[i >> 9] + 1e-9f);   // H == 512 == 1<<9
    __bf16 h = (__bf16)m;
    hi[i] = h;
    lo[i] = (__bf16)(m - (float)h);
}

__global__ void k_split(const float* __restrict__ src,
                        __bf16* __restrict__ hi, __bf16* __restrict__ lo, int n) {
    int i = blockIdx.x * blockDim.x + threadIdx.x;
    if (i >= n) return;
    float f = src[i];
    __bf16 h = (__bf16)f;
    hi[i] = h;
    lo[i] = (__bf16)(f - (float)h);
}

// ---------------- phase 4: fused split-bf16 WMMA GEMM + GRU gates ----------
// A-frag (16x32 bf16): lane holds row M=lane%16, two 16B runs at K = kb and kb+16,
// kb = 8*(lane/16).  B-frag (32x16): lane holds col N=lane%16 of B = row (c0+N) of W,
// same K packing.  C/D f32: elem r -> row 16*kt + r + kb, col c0 + lane%16.
__device__ __forceinline__ v16bf load_frag(const __bf16* rowbase, int k0) {
    union { uint4 u[2]; v16bf v; } u;
    u.u[0] = *reinterpret_cast<const uint4*>(rowbase + k0);
    u.u[1] = *reinterpret_cast<const uint4*>(rowbase + k0 + 16);
    return u.v;
}

#define WMMA_BF16(A, Bm, C) \
    __builtin_amdgcn_wmma_f32_16x16x32_bf16(false, (A), false, (Bm), (short)0, (C), false, false)

__global__ void __launch_bounds__(128)
k_gru(const __bf16* __restrict__ mh,    const __bf16* __restrict__ ml,
      const __bf16* __restrict__ ah,    const __bf16* __restrict__ al,
      const __bf16* __restrict__ wih_h, const __bf16* __restrict__ wih_l,
      const __bf16* __restrict__ whh_h, const __bf16* __restrict__ whh_l,
      const float*  __restrict__ A_states, const float* __restrict__ cnt,
      const float*  __restrict__ b_ih,  const float* __restrict__ b_hh,
      float* __restrict__ out) {
    const int lane = threadIdx.x & 31;
    const int wave = threadIdx.x >> 5;
    const int n    = lane & 15;
    const int kb   = (lane >> 4) << 3;            // 0 or 8
    const int kt   = blockIdx.x;                  // 0..K/16-1
    const int c0   = (blockIdx.y * 4 + wave) * 16; // column tile in [0,H)

    const __bf16* pmh = mh + (size_t)(kt * 16 + n) * HH;
    const __bf16* pml = ml + (size_t)(kt * 16 + n) * HH;
    const __bf16* pah = ah + (size_t)(kt * 16 + n) * HH;
    const __bf16* pal = al + (size_t)(kt * 16 + n) * HH;

    const __bf16* pih_h[3]; const __bf16* pih_l[3];
    const __bf16* phh_h[3]; const __bf16* phh_l[3];
#pragma unroll
    for (int g = 0; g < 3; ++g) {
        size_t r = (size_t)(g * HH + c0 + n) * HH;
        pih_h[g] = wih_h + r;  pih_l[g] = wih_l + r;
        phh_h[g] = whh_h + r;  phh_l[g] = whh_l + r;
    }

    v8f acc_i[3] = {};
    v8f acc_h[3] = {};

    for (int kk = 0; kk < HH; kk += 32) {
        const int ko = kk + kb;
        v16bf a_mh = load_frag(pmh, ko);
        v16bf a_ml = load_frag(pml, ko);
        v16bf a_ah = load_frag(pah, ko);
        v16bf a_al = load_frag(pal, ko);
#pragma unroll
        for (int g = 0; g < 3; ++g) {
            v16bf b_h = load_frag(pih_h[g], ko);
            v16bf b_l = load_frag(pih_l[g], ko);
            acc_i[g] = WMMA_BF16(a_mh, b_h, acc_i[g]);
            acc_i[g] = WMMA_BF16(a_mh, b_l, acc_i[g]);
            acc_i[g] = WMMA_BF16(a_ml, b_h, acc_i[g]);
            b_h = load_frag(phh_h[g], ko);
            b_l = load_frag(phh_l[g], ko);
            acc_h[g] = WMMA_BF16(a_ah, b_h, acc_h[g]);
            acc_h[g] = WMMA_BF16(a_ah, b_l, acc_h[g]);
            acc_h[g] = WMMA_BF16(a_al, b_h, acc_h[g]);
        }
    }

    float bi[3], bh2[3];
#pragma unroll
    for (int g = 0; g < 3; ++g) {
        bi[g]  = b_ih[g * HH + c0 + n];
        bh2[g] = b_hh[g * HH + c0 + n];
    }

#pragma unroll
    for (int r = 0; r < 8; ++r) {
        int row = kt * 16 + r + kb;
        size_t off = (size_t)row * HH + c0 + n;
        float a_st = A_states[off];
        float ir = acc_i[0][r] + bi[0];
        float iz = acc_i[1][r] + bi[1];
        float in_ = acc_i[2][r] + bi[2];
        float hr = acc_h[0][r] + bh2[0];
        float hz = acc_h[1][r] + bh2[1];
        float hn = acc_h[2][r] + bh2[2];
        float rg = 1.0f / (1.0f + expf(-(ir + hr)));
        float zg = 1.0f / (1.0f + expf(-(iz + hz)));
        float ng = tanhf(in_ + rg * hn);
        float nv = (1.0f - zg) * ng + zg * a_st;
        float res = 0.5f * a_st + 0.5f * nv;     // ema_m = 0.5
        out[off] = (cnt[row] > 0.0f) ? res : a_st;
    }
}

// ---------------------------------------------------------------------------
extern "C" void kernel_launch(void* const* d_in, const int* in_sizes, int n_in,
                              void* d_out, int out_size, void* d_ws, size_t ws_size,
                              hipStream_t stream) {
    (void)in_sizes; (void)n_in; (void)out_size; (void)ws_size;
    const float*     h_fast   = (const float*)d_in[0];
    const float*     h_slow   = (const float*)d_in[1];
    const long long* idx      = (const long long*)d_in[2];
    const float*     A_states = (const float*)d_in[3];
    const float*     W_ih     = (const float*)d_in[4];
    const float*     W_hh     = (const float*)d_in[5];
    const float*     b_ih     = (const float*)d_in[6];
    const float*     b_hh     = (const float*)d_in[7];
    float*           out      = (float*)d_out;

    // workspace layout
    float*  sums  = (float*)d_ws;                          // K*H f32
    float*  cnt   = sums + (size_t)KK * HH;                // K   f32
    __bf16* mh    = (__bf16*)(cnt + KK);                   // K*H bf16
    __bf16* ml    = mh + (size_t)KK * HH;
    __bf16* ah    = ml + (size_t)KK * HH;
    __bf16* al    = ah + (size_t)KK * HH;
    __bf16* wih_h = al + (size_t)KK * HH;                  // 3H*H bf16
    __bf16* wih_l = wih_h + (size_t)G3H * HH;
    __bf16* whh_h = wih_l + (size_t)G3H * HH;
    __bf16* whh_l = whh_h + (size_t)G3H * HH;

    int n4 = (KK * HH + KK) / 4;
    k_zero<<<(n4 + 255) / 256, 256, 0, stream>>>((float4*)sums, n4);

    k_scatter<<<BB, DD, 0, stream>>>(h_fast, h_slow, idx, sums, cnt);

    int nKH = KK * HH;
    k_mean_split<<<(nKH + 255) / 256, 256, 0, stream>>>(sums, cnt, mh, ml, nKH);
    k_split<<<(nKH + 255) / 256, 256, 0, stream>>>(A_states, ah, al, nKH);
    int nW = G3H * HH;
    k_split<<<(nW + 255) / 256, 256, 0, stream>>>(W_ih, wih_h, wih_l, nW);
    k_split<<<(nW + 255) / 256, 256, 0, stream>>>(W_hh, whh_h, whh_l, nW);

    dim3 grid(KK / 16, HH / 64);    // (512, 8), 4 waves/block = 4 column tiles
    k_gru<<<grid, 128, 0, stream>>>(mh, ml, ah, al, wih_h, wih_l, whh_h, whh_l,
                                    A_states, cnt, b_ih, b_hh, out);
}
